// DebertaCRFBoundaryDetector_86998857548523
// MI455X (gfx1250) — matrix operand order repaired and checked by reading, not compiled, not run on hardware
//
#include <hip/hip_runtime.h>
#include <hip/hip_bf16.h>
#include <math.h>

// Problem constants (from reference)
#define B_DIM 16
#define S_DIM 2048
#define H_DIM 1024
#define L_DIM 2
#define NPAD 16                  // WMMA N dimension (L padded 2 -> 16)
#define M_ROWS (B_DIM * S_DIM)   // 32768 output rows of the emissions GEMM
#define NTILES (M_ROWS / 16)     // 2048 16-row WMMA tiles
#define WAVES_PER_BLOCK 8
#define CHUNK_LEN 128
#define NCHUNKS 16

#define ALPHA_CRF_V 1.0f
#define ALPHA_CE_V  1.0f
#define ALPHA_TV_V  0.5f
#define SHORT_ALPHA_V 2.0f
#define EPS_V 1e-8f

typedef _Float16 v16h __attribute__((ext_vector_type(16)));
typedef float v8f  __attribute__((ext_vector_type(8)));
typedef float v4f  __attribute__((ext_vector_type(4)));
typedef float v2f  __attribute__((ext_vector_type(2)));

// Workspace layout (float offsets)
#define OFF_E      0                                  // 65536 floats: emissions (B*S, L)
#define OFF_CHUNK  (M_ROWS * L_DIM)                   // 1024 floats: 16x16 chunk 2x2 matrices
#define OFF_NUM    (OFF_CHUNK + B_DIM * NCHUNKS * 4)  // 16
#define OFF_CE     (OFF_NUM + B_DIM)                  // 16
#define OFF_MS     (OFF_CE + B_DIM)                   // 16
#define OFF_TV     (OFF_MS + B_DIM)                   // 16
#define OFF_VM     (OFF_TV + B_DIM)                   // 16

__device__ __forceinline__ float logadd2(float a, float b) {
  float mx = fmaxf(a, b), mn = fminf(a, b);
  return mx + log1pf(__expf(mn - mx));
}

// ---------------------------------------------------------------------------
// Kernel 1: emissions = hidden @ W^T + b, via v_wmma_f32_16x16x32_f16.
// One wave per 16-row tile; N padded 2 -> 16; K loop 1024/32 = 32 WMMAs.
// W is staged into LDS as f16 padded to all 16 N-rows (rows >= 2 are zero),
// so the inner loop is completely branch-free (no EXEC manipulation, no
// per-iteration zeroing). HBM-bound: single 128 MB pass of `hidden`.
// ---------------------------------------------------------------------------
__global__ __launch_bounds__(256) void k_emissions(const float* __restrict__ hidden,
                                                   const float* __restrict__ Wg,
                                                   const float* __restrict__ bias,
                                                   float* __restrict__ Eout) {
  __shared__ v16h WldsV[NPAD * H_DIM / 16];  // 16 padded rows x 1024 halves = 32 KB
  _Float16* Wh = (_Float16*)WldsV;
  int tid = threadIdx.x;
  for (int i = tid; i < NPAD * H_DIM; i += 256)
    Wh[i] = (i < L_DIM * H_DIM) ? (_Float16)Wg[i] : (_Float16)0.0f;
  __syncthreads();

  int lane = tid & 31;
  int wave = tid >> 5;
  int tile = blockIdx.x * WAVES_PER_BLOCK + wave;
  int row0 = tile * 16;
  int m    = row0 + (lane & 15);
  const float* hrow = hidden + (size_t)m * H_DIM;
  bool lo = lane < 16;
  int n = lane & 15;   // padded N column held by this lane (B/C/D layouts)

  v8f acc = {};
  for (int k0 = 0; k0 < H_DIM; k0 += 32) {
    // A fragment (contiguous K, matching B): lanes 0-15 -> K k0+0..15,
    // lanes 16-31 -> K k0+16..31.
    int ka = k0 + (lo ? 0 : 16);
    v4f a0 = *(const v4f*)(hrow + ka);
    v4f a1 = *(const v4f*)(hrow + ka + 4);
    v4f a2 = *(const v4f*)(hrow + ka + 8);
    v4f a3 = *(const v4f*)(hrow + ka + 12);
    v16h A;
#pragma unroll
    for (int e = 0; e < 4; ++e) {
      A[e]      = (_Float16)a0[e];
      A[4 + e]  = (_Float16)a1[e];
      A[8 + e]  = (_Float16)a2[e];
      A[12 + e] = (_Float16)a3[e];
    }
    // B fragment (contiguous K, matching A): lane holds column n; halves are
    // K = k0 + e (lanes 0-15) or k0 + 16 + e (lanes 16-31). B[k][n] = W[n][k],
    // with W rows >= L_DIM zero-padded in LDS -> unconditional load.
    int kb16 = (k0 + (lo ? 0 : 16)) >> 4;
    v16h Bf = WldsV[n * (H_DIM / 16) + kb16];
    acc = __builtin_amdgcn_wmma_f32_16x16x32_f16(false, A, false, Bf,
                                                 (short)0, acc, false, false);
  }
  // D layout: VGPR v -> row M = v (lanes 0-15) or v+8 (lanes 16-31), col = lane&15
  if (n < L_DIM) {
    float bn = bias[n];
    int rbase = row0 + (lo ? 0 : 8);
#pragma unroll
    for (int v = 0; v < 8; ++v)
      Eout[(size_t)(rbase + v) * L_DIM + n] = acc[v] + bn;
  }
}

// ---------------------------------------------------------------------------
// Kernel 2: CRF forward scan, parallelized as chunked 2x2 log-semiring
// matrix products. 16 batches x 16 chunks = 256 independent threads.
// ---------------------------------------------------------------------------
__global__ void k_crf_chunks(const float* __restrict__ E,
                             const float* __restrict__ trans,
                             float* __restrict__ chunkM) {
  int tid = threadIdx.x;            // 256 threads
  int b = tid >> 4, c = tid & 15;
  const v2f* Eb = (const v2f*)(E + (size_t)b * S_DIM * L_DIM);
  float t00 = trans[0], t01 = trans[1], t10 = trans[2], t11 = trans[3];
  const float NEG = -1e30f;
  float m00 = 0.f, m01 = NEG, m10 = NEG, m11 = 0.f;   // log-semiring identity
  int ts = 1 + c * CHUNK_LEN;
  int te = min(S_DIM, ts + CHUNK_LEN);
  for (int t = ts; t < te; ++t) {
    v2f e = Eb[t];
    // T_t[i][j] = trans[i][j] + em_t[j];  M <- M (x) T_t
    float n00 = e[0] + logadd2(m00 + t00, m01 + t10);
    float n01 = e[1] + logadd2(m00 + t01, m01 + t11);
    float n10 = e[0] + logadd2(m10 + t00, m11 + t10);
    float n11 = e[1] + logadd2(m10 + t01, m11 + t11);
    m00 = n00; m01 = n01; m10 = n10; m11 = n11;
  }
  float* o = chunkM + (size_t)(b * NCHUNKS + c) * 4;
  o[0] = m00; o[1] = m01; o[2] = m10; o[3] = m11;
}

// ---------------------------------------------------------------------------
// Kernel 3: per batch row — weighted CE (run-length weights via block scan +
// LDS atomic histogram), CRF numerator partial (tag + transition scores),
// mask sum, and TV penalty partials. One 256-thread workgroup per row.
// ---------------------------------------------------------------------------
__global__ __launch_bounds__(256) void k_row_stats(const float* __restrict__ E,
                                                   const int* __restrict__ labels,
                                                   const int* __restrict__ mask,
                                                   const float* __restrict__ trans,
                                                   float* __restrict__ ws) {
  __shared__ int   lab[S_DIM];
  __shared__ int   seg[S_DIM];
  __shared__ int   lens[S_DIM];
  __shared__ float mf[S_DIM];
  __shared__ float red[256];
  __shared__ int   totals[256];
  int b = blockIdx.x, tid = threadIdx.x;
  const int* labg = labels + (size_t)b * S_DIM;
  const int* mg   = mask   + (size_t)b * S_DIM;
  for (int i = tid; i < S_DIM; i += 256) { lab[i] = labg[i]; mf[i] = (float)mg[i]; lens[i] = 0; }
  __syncthreads();

  // segment ids: cumsum(change) - 1 (two-level scan: 8 local + 256 serial)
  int base = tid * 8;
  int run = 0;
#pragma unroll
  for (int j = 0; j < 8; ++j) {
    int t = base + j;
    int ch = (t == 0) ? 1 : (lab[t] != lab[t - 1]);
    run += ch;
    seg[t] = run;
  }
  totals[tid] = run;
  __syncthreads();
  if (tid == 0) {
    int s = 0;
    for (int i = 0; i < 256; ++i) { int v = totals[i]; totals[i] = s; s += v; }
  }
  __syncthreads();
  int off = totals[tid] - 1;
#pragma unroll
  for (int j = 0; j < 8; ++j) seg[base + j] += off;
  __syncthreads();
#pragma unroll
  for (int j = 0; j < 8; ++j) atomicAdd(&lens[seg[base + j]], 1);
  __syncthreads();

  float t00 = trans[0], t01 = trans[1], t10 = trans[2], t11 = trans[3];
  const float* Eb = E + (size_t)b * S_DIM * L_DIM;
  float ce_acc = 0.f, num_acc = 0.f, ms_acc = 0.f, tv_acc = 0.f, vm_acc = 0.f;
  for (int j = 0; j < 8; ++j) {
    int t = base + j;
    float e0 = Eb[2 * t], e1 = Eb[2 * t + 1];
    float mx  = fmaxf(e0, e1);
    float lse = mx + __logf(__expf(e0 - mx) + __expf(e1 - mx));
    int l = lab[t];
    float el  = l ? e1 : e0;
    float mft = mf[t];
    float tl  = (float)lens[seg[t]];
    float w   = 1.0f + SHORT_ALPHA_V * rsqrtf(tl);
    ce_acc += (lse - el) * w * mft;    // ce = -log_softmax[label]
    ms_acc += mft;
    if (t == 0) {
      num_acc += el;                   // tag score at t=0 (start/end added later)
    } else {
      int lp = lab[t - 1];
      float tr = lp ? (l ? t11 : t10) : (l ? t01 : t00);
      num_acc += (tr + el) * mft;
      // TV penalty term
      float p0 = __expf(e0 - lse), p1 = __expf(e1 - lse);
      float f0 = Eb[2 * t - 2], f1 = Eb[2 * t - 1];
      float mx2  = fmaxf(f0, f1);
      float lse2 = mx2 + __logf(__expf(f0 - mx2) + __expf(f1 - mx2));
      float q0 = __expf(f0 - lse2), q1 = __expf(f1 - lse2);
      float vm = mft * mf[t - 1];
      tv_acc += (fabsf(p0 - q0) + fabsf(p1 - q1)) * vm;
      vm_acc += vm;
    }
  }

  float vals[5] = {ce_acc, num_acc, ms_acc, tv_acc, vm_acc};
  int   offs[5] = {OFF_CE, OFF_NUM, OFF_MS, OFF_TV, OFF_VM};
  for (int q = 0; q < 5; ++q) {
    red[tid] = vals[q];
    __syncthreads();
    for (int s = 128; s > 0; s >>= 1) {
      if (tid < s) red[tid] += red[tid + s];
      __syncthreads();
    }
    if (tid == 0) ws[offs[q] + b] = red[0];
    __syncthreads();
  }
}

// ---------------------------------------------------------------------------
// Kernel 4: fold chunk matrices -> logZ per batch, finish numerator with
// start/end terms, and assemble the scalar loss.
// ---------------------------------------------------------------------------
__global__ void k_combine(const float* __restrict__ ws_all,
                          const float* __restrict__ startT,
                          const float* __restrict__ endT,
                          const int* __restrict__ labels,
                          float* __restrict__ out) {
  __shared__ float s_llh[16], s_ce[16], s_ms[16], s_tv[16], s_vm[16];
  int b = threadIdx.x;  // 16 threads
  const float* E      = ws_all + OFF_E;
  const float* chunkM = ws_all + OFF_CHUNK;
  const v2f* Eb = (const v2f*)(E + (size_t)b * S_DIM * L_DIM);
  v2f e0 = Eb[0];
  float a0 = startT[0] + e0[0], a1 = startT[1] + e0[1];
  for (int c = 0; c < NCHUNKS; ++c) {
    const float* M = chunkM + (size_t)(b * NCHUNKS + c) * 4;
    float m00 = M[0], m01 = M[1], m10 = M[2], m11 = M[3];
    float na0 = logadd2(a0 + m00, a1 + m10);
    float na1 = logadd2(a0 + m01, a1 + m11);
    a0 = na0; a1 = na1;
  }
  float logZ = logadd2(a0 + endT[0], a1 + endT[1]);
  float msum = ws_all[OFF_MS + b];
  int last = (int)(msum + 0.5f) - 1;
  if (last < 0) last = 0;
  int l0 = labels[(size_t)b * S_DIM];
  int ll = labels[(size_t)b * S_DIM + last];
  float num = ws_all[OFF_NUM + b] + startT[l0] + endT[ll];
  s_llh[b] = num - logZ;
  s_ce[b]  = ws_all[OFF_CE + b];
  s_ms[b]  = msum;
  s_tv[b]  = ws_all[OFF_TV + b];
  s_vm[b]  = ws_all[OFF_VM + b];
  __syncthreads();
  if (b == 0) {
    float llh = 0, ce = 0, ms = 0, tv = 0, vm = 0;
    for (int i = 0; i < 16; ++i) {
      llh += s_llh[i]; ce += s_ce[i]; ms += s_ms[i]; tv += s_tv[i]; vm += s_vm[i];
    }
    float loss = -ALPHA_CRF_V * (llh / ms)
               +  ALPHA_CE_V  * (ce / (ms + EPS_V))
               +  ALPHA_TV_V  * (tv / (vm + EPS_V));
    out[0] = loss;
  }
}

extern "C" void kernel_launch(void* const* d_in, const int* in_sizes, int n_in,
                              void* d_out, int out_size, void* d_ws, size_t ws_size,
                              hipStream_t stream) {
  const float* hidden = (const float*)d_in[0];
  const float* W      = (const float*)d_in[1];
  const float* bias   = (const float*)d_in[2];
  const float* startT = (const float*)d_in[3];
  const float* endT   = (const float*)d_in[4];
  const float* trans  = (const float*)d_in[5];
  const int*   labels = (const int*)d_in[6];
  const int*   mask   = (const int*)d_in[7];
  float* ws  = (float*)d_ws;
  float* out = (float*)d_out;

  k_emissions<<<NTILES / WAVES_PER_BLOCK, 256, 0, stream>>>(hidden, W, bias, ws + OFF_E);
  k_crf_chunks<<<1, 256, 0, stream>>>(ws + OFF_E, trans, ws + OFF_CHUNK);
  k_row_stats<<<B_DIM, 256, 0, stream>>>(ws + OFF_E, labels, mask, trans, ws);
  k_combine<<<1, 16, 0, stream>>>(ws, startT, endT, labels, out);
}
